// Head_9483287789886
// MI455X (gfx1250) — matrix operand order repaired and checked by reading, not compiled, not run on hardware
//
#include <hip/hip_runtime.h>
#include <hip/hip_bf16.h>

// Causal attention head, B=8 T=2048 C=384 H=64, softmax over QUERY axis
// (axis=1) per the reference. Column-stats two-pass formulation.
// All matmuls: bf16x3 (hi/lo split) WMMA, fp32 accumulate.
// Precision-splitting happens ONCE (prep kernels / projection stores);
// the hot attention loops load bf16 fragments directly (no cvt VALU).
//
// ws layout (~12.9 MB):
//   qh,ql,kh,kl : bf16 (B*T, H) row-major        4 * 2MB
//   vth,vtl     : bf16 (B*H, T) transposed       2 * 2MB
//   colmax,colsum : f32 (B*T)                    2 * 64KB
//   wth,wtl     : bf16 (3*H, C) transposed W     2 * 144KB

typedef __attribute__((ext_vector_type(16))) __bf16 v16bf;
typedef __attribute__((ext_vector_type(8)))  __bf16 v8bf;
typedef __attribute__((ext_vector_type(8)))  float  v8f;

#define Bc 8
#define Tc 2048
#define Cc 384
#define Hc 64
#define NTILES_T (Tc / 16)          // 128
#define SCALE 0.05103103630798288f  // 384^-0.5

// ISA 7.12.2: 16-bit A-matrix 16x32 per-lane K index for vector element j
__device__ __forceinline__ int a_kmap(int j, int hi) {
  return hi * 8 + j + ((j >= 8) ? 8 : 0);
}

// A-fragment: two contiguous 8-elem chunks at K = hi*8 and K = 16 + hi*8
__device__ __forceinline__ v16bf ld_afrag(const __bf16* slab, int hi) {
  v8bf c0 = *reinterpret_cast<const v8bf*>(slab + hi * 8);
  v8bf c1 = *reinterpret_cast<const v8bf*>(slab + 16 + hi * 8);
  return __builtin_shufflevector(c0, c1, 0, 1, 2, 3, 4, 5, 6, 7, 8, 9, 10, 11,
                                 12, 13, 14, 15);
}
// B-fragment: two contiguous 8-elem chunks at K = hi*8 and K = 16 + hi*8
__device__ __forceinline__ v16bf ld_bfrag(const __bf16* slab, int hi) {
  v8bf c0 = *reinterpret_cast<const v8bf*>(slab + hi * 8);
  v8bf c1 = *reinterpret_cast<const v8bf*>(slab + 16 + hi * 8);
  return __builtin_shufflevector(c0, c1, 0, 1, 2, 3, 4, 5, 6, 7, 8, 9, 10, 11,
                                 12, 13, 14, 15);
}

__device__ __forceinline__ v8f wmma_bf16x3(v16bf ah, v16bf al, v16bf bh,
                                           v16bf bl, v8f c) {
  c = __builtin_amdgcn_wmma_f32_16x16x32_bf16(false, ah, false, bh, (short)0, c,
                                              false, false);
  c = __builtin_amdgcn_wmma_f32_16x16x32_bf16(false, ah, false, bl, (short)0, c,
                                              false, false);
  c = __builtin_amdgcn_wmma_f32_16x16x32_bf16(false, al, false, bh, (short)0, c,
                                              false, false);
  return c;
}

__device__ __forceinline__ void split_bf(float v, __bf16& h, __bf16& l) {
  h = (__bf16)v;
  l = (__bf16)(v - (float)h);
}

__device__ __forceinline__ unsigned int pack_bf2(float w) {
  __bf16 h, l;
  split_bf(w, h, l);
  union { __bf16 b; unsigned short u; } ch, cl;
  ch.b = h;
  cl.b = l;
  return (unsigned int)ch.u | ((unsigned int)cl.u << 16);
}
__device__ __forceinline__ void unpack_bf2(unsigned int p, __bf16& h,
                                           __bf16& l) {
  union { unsigned short u; __bf16 b; } ch, cl;
  ch.u = (unsigned short)(p & 0xffffu);
  cl.u = (unsigned short)(p >> 16);
  h = ch.b;
  l = cl.b;
}

// ---------------------------------------------------------------------------
// Kernel 0: transpose + hi/lo split the three weights into (3*H, C) bf16.
// ---------------------------------------------------------------------------
__global__ __launch_bounds__(256) void prep_w(const float* __restrict__ Wq,
                                              const float* __restrict__ Wk,
                                              const float* __restrict__ Wv,
                                              __bf16* __restrict__ wth,
                                              __bf16* __restrict__ wtl) {
  int idx = blockIdx.x * 256 + threadIdx.x;
  if (idx >= 3 * Cc * Hc) return;
  int w = idx / (Cc * Hc);
  int rem = idx - w * (Cc * Hc);
  int c = rem / Hc;
  int h = rem - c * Hc;
  const float* W = (w == 0) ? Wq : (w == 1) ? Wk : Wv;
  __bf16 bh, bl;
  split_bf(W[rem], bh, bl);
  size_t o = ((size_t)w * Hc + h) * Cc + c;
  wth[o] = bh;
  wtl[o] = bl;
}

// ---------------------------------------------------------------------------
// Kernel 1: fused QKV projection. Wave: 16 rows x all 12 (3 W x 4 h) tiles.
// Outputs bf16 hi/lo: q,k row-major (T,H); v transposed (H,T).
// ---------------------------------------------------------------------------
__global__ __launch_bounds__(256) void qkv_proj(
    const float* __restrict__ x, const __bf16* __restrict__ wth,
    const __bf16* __restrict__ wtl, __bf16* __restrict__ qh,
    __bf16* __restrict__ ql, __bf16* __restrict__ kh,
    __bf16* __restrict__ kl, __bf16* __restrict__ vth,
    __bf16* __restrict__ vtl) {
  const int lane = threadIdx.x & 31;
  const int wave = threadIdx.x >> 5;
  const int l15 = lane & 15;
  const int hi = lane >> 4;
  const int rowBase = (blockIdx.x * 8 + wave) * 16;

  v8f acc[12];
#pragma unroll
  for (int t = 0; t < 12; ++t)
#pragma unroll
    for (int e = 0; e < 8; ++e) acc[t][e] = 0.f;

  for (int kk = 0; kk < Cc / 32; ++kk) {
    if (kk + 1 < Cc / 32)  // CDNA5 global_prefetch_b8 of next K slab
      __builtin_prefetch(&x[(size_t)(rowBase + l15) * Cc + (kk + 1) * 32], 0, 0);
    // A fragment (x rows): the only per-iteration fp32->bf16 split.
    v16bf ah, al;
#pragma unroll
    for (int j = 0; j < 16; ++j) {
      float xv = x[(size_t)(rowBase + l15) * Cc + kk * 32 + a_kmap(j, hi)];
      __bf16 h2, l2;
      split_bf(xv, h2, l2);
      ah[j] = h2;
      al[j] = l2;
    }
#pragma unroll
    for (int wt = 0; wt < 12; ++wt) {  // wt = w*4 + ht
      const size_t wrow = ((size_t)wt * 16 + l15) * Cc + kk * 32;
      v16bf bh = ld_bfrag(wth + wrow, hi);
      v16bf bl = ld_bfrag(wtl + wrow, hi);
      acc[wt] = wmma_bf16x3(ah, al, bh, bl, acc[wt]);
    }
  }
  // store with one-time hi/lo split
#pragma unroll
  for (int wt = 0; wt < 12; ++wt) {
    const int w = wt >> 2;
    const int hcol = (wt & 3) * 16 + l15;  // lane owns column hcol
#pragma unroll
    for (int r = 0; r < 8; ++r) {
      const int row = rowBase + r + hi * 8;
      __bf16 h2, l2;
      split_bf(acc[wt][r], h2, l2);
      if (w == 0) {
        qh[(size_t)row * Hc + hcol] = h2;
        ql[(size_t)row * Hc + hcol] = l2;
      } else if (w == 1) {
        kh[(size_t)row * Hc + hcol] = h2;
        kl[(size_t)row * Hc + hcol] = l2;
      } else {  // v: transposed (B*H, T)
        const int b = row >> 11;
        const int tloc = row & (Tc - 1);
        vth[((size_t)b * Hc + hcol) * Tc + tloc] = h2;
        vtl[((size_t)b * Hc + hcol) * Tc + tloc] = l2;
      }
    }
  }
}

// ---------------------------------------------------------------------------
// Kernel 2: per-column softmax stats (softmax over query axis).
// Wave owns one (b, sTile); streams tTiles >= sTile with online max/sum.
// ---------------------------------------------------------------------------
__global__ __launch_bounds__(256) void col_stats(
    const __bf16* __restrict__ qh, const __bf16* __restrict__ ql,
    const __bf16* __restrict__ kh, const __bf16* __restrict__ kl,
    float* __restrict__ colmax, float* __restrict__ colsum) {
  const int lane = threadIdx.x & 31;
  const int wave = threadIdx.x >> 5;
  const int l15 = lane & 15;
  const int hi = lane >> 4;
  const int job = blockIdx.x * 8 + wave;  // 8 * 128 = 1024 jobs
  const int b = job >> 7;
  const int sTile = job & 127;
  const int sBase = sTile * 16;

  // K^T B-fragments (K dim = h, 2 steps of 32): contiguous bf16 loads
  const size_t krow = ((size_t)b * Tc + sBase + l15) * Hc;
  v16bf kfh[2], kfl[2];
#pragma unroll
  for (int st = 0; st < 2; ++st) {
    kfh[st] = ld_bfrag(kh + krow + st * 32, hi);
    kfl[st] = ld_bfrag(kl + krow + st * 32, hi);
  }

  float mrun = -__builtin_inff();
  float drun = 0.f;

  for (int tTile = sTile; tTile < NTILES_T; ++tTile) {
    const int tBase = tTile * 16;
    const size_t qrow = ((size_t)b * Tc + tBase + l15) * Hc;
    v8f c;
#pragma unroll
    for (int e = 0; e < 8; ++e) c[e] = 0.f;
#pragma unroll
    for (int st = 0; st < 2; ++st) {
      v16bf afh = ld_afrag(qh + qrow + st * 32, hi);
      v16bf afl = ld_afrag(ql + qrow + st * 32, hi);
      c = wmma_bf16x3(afh, afl, kfh[st], kfl[st], c);
    }
    const int s = sBase + l15;
#pragma unroll
    for (int r = 0; r < 8; ++r) {
      int t = tBase + r + hi * 8;
      if (t >= s) {  // only the diagonal tile ever masks
        float sc = c[r] * SCALE;
        float mn = fmaxf(mrun, sc);
        drun = drun * __expf(mrun - mn) + __expf(sc - mn);
        mrun = mn;
      }
    }
  }
  // combine the two half-wave row groups
  float m2 = __shfl_xor(mrun, 16, 32);
  float d2 = __shfl_xor(drun, 16, 32);
  float M = fmaxf(mrun, m2);
  float D = drun * __expf(mrun - M) + d2 * __expf(m2 - M);
  if (hi == 0) {
    colmax[(size_t)b * Tc + sBase + l15] = M;
    colsum[(size_t)b * Tc + sBase + l15] = D;
  }
}

// ---------------------------------------------------------------------------
// Kernel 3: output. Wave owns (b, tTile); recomputes score tiles in sTile
// pairs, LDS-transposes packed hi/lo probabilities into a 16x32 A-fragment,
// accumulates W @ V from transposed V (contiguous B-fragments).
// ---------------------------------------------------------------------------
__global__ __launch_bounds__(256) void attn_out(
    const __bf16* __restrict__ qh, const __bf16* __restrict__ ql,
    const __bf16* __restrict__ kh, const __bf16* __restrict__ kl,
    const __bf16* __restrict__ vth, const __bf16* __restrict__ vtl,
    const float* __restrict__ colmax, const float* __restrict__ colsum,
    float* __restrict__ out) {
  __shared__ unsigned int ldsW[8 * 16 * 32];  // per-wave packed (hi,lo) tile
  const int lane = threadIdx.x & 31;
  const int wave = threadIdx.x >> 5;
  const int l15 = lane & 15;
  const int hi = lane >> 4;
  const int job = blockIdx.x * 8 + wave;
  const int b = job >> 7;
  const int tTile = job & 127;
  const int tBase = tTile * 16;
  unsigned int* lw = &ldsW[wave * 512];

  // Q A-fragments, loaded once
  const size_t qrow = ((size_t)b * Tc + tBase + l15) * Hc;
  v16bf afh[2], afl[2];
#pragma unroll
  for (int st = 0; st < 2; ++st) {
    afh[st] = ld_afrag(qh + qrow + st * 32, hi);
    afl[st] = ld_afrag(ql + qrow + st * 32, hi);
  }

  v8f o[4];
#pragma unroll
  for (int ht = 0; ht < 4; ++ht)
#pragma unroll
    for (int e = 0; e < 8; ++e) o[ht][e] = 0.f;

  for (int sp = 0; sp <= tTile / 2; ++sp) {
    const int sTile0 = 2 * sp;
#pragma unroll
    for (int st = 0; st < 2; ++st) {
      const int sT = sTile0 + st;
      if (sT <= tTile) {
        const int sBase = sT * 16;
        const size_t krow = ((size_t)b * Tc + sBase + l15) * Hc;
        v8f c;
#pragma unroll
        for (int e = 0; e < 8; ++e) c[e] = 0.f;
#pragma unroll
        for (int ks = 0; ks < 2; ++ks) {
          v16bf bh = ld_bfrag(kh + krow + ks * 32, hi);
          v16bf bl = ld_bfrag(kl + krow + ks * 32, hi);
          c = wmma_bf16x3(afh[ks], afl[ks], bh, bl, c);
        }
        const int s = sBase + l15;
        const float ms = colmax[(size_t)b * Tc + s];
        const float rs = 1.0f / colsum[(size_t)b * Tc + s];
#pragma unroll
        for (int r = 0; r < 8; ++r) {
          int t = tBase + r + hi * 8;
          float wv = (t >= s) ? __expf(c[r] * SCALE - ms) * rs : 0.f;
          lw[(r + hi * 8) * 32 + st * 16 + l15] = pack_bf2(wv);
        }
      } else {
#pragma unroll
        for (int r = 0; r < 8; ++r) lw[(r + hi * 8) * 32 + st * 16 + l15] = 0u;
      }
    }
    // Read probabilities back in A-fragment layout (4x ds_load_b128).
    v16bf wh, wl;
    {
      const uint4* lp0 =
          reinterpret_cast<const uint4*>(lw + l15 * 32 + hi * 8);
      const uint4* lp1 =
          reinterpret_cast<const uint4*>(lw + l15 * 32 + 16 + hi * 8);
      uint4 p0 = lp0[0], p1 = lp0[1], p2 = lp1[0], p3 = lp1[1];
      unsigned int ps[16] = {p0.x, p0.y, p0.z, p0.w, p1.x, p1.y, p1.z, p1.w,
                             p2.x, p2.y, p2.z, p2.w, p3.x, p3.y, p3.z, p3.w};
#pragma unroll
      for (int j = 0; j < 16; ++j) {
        __bf16 h2, l2;
        unpack_bf2(ps[j], h2, l2);
        wh[j] = h2;
        wl[j] = l2;
      }
    }
    // V B-fragments from transposed V: contiguous bf16 loads.
    const int sBase0 = sTile0 * 16;
#pragma unroll
    for (int ht = 0; ht < 4; ++ht) {
      const size_t vrow =
          ((size_t)b * Hc + ht * 16 + l15) * Tc + sBase0;
      v16bf vh = ld_bfrag(vth + vrow, hi);
      v16bf vl = ld_bfrag(vtl + vrow, hi);
      o[ht] = wmma_bf16x3(wh, wl, vh, vl, o[ht]);
    }
  }
#pragma unroll
  for (int ht = 0; ht < 4; ++ht)
#pragma unroll
    for (int r = 0; r < 8; ++r)
      out[((size_t)b * Tc + tBase + r + hi * 8) * Hc + ht * 16 + l15] =
          o[ht][r];
}

extern "C" void kernel_launch(void* const* d_in, const int* in_sizes, int n_in,
                              void* d_out, int out_size, void* d_ws,
                              size_t ws_size, hipStream_t stream) {
  (void)in_sizes; (void)n_in; (void)out_size; (void)ws_size;
  const float* x = (const float*)d_in[0];
  const float* Wq = (const float*)d_in[1];
  const float* Wk = (const float*)d_in[2];
  const float* Wv = (const float*)d_in[3];
  float* out = (float*)d_out;

  const size_t MH = (size_t)Bc * Tc * Hc;  // 1,048,576
  char* p = (char*)d_ws;
  __bf16* qh = (__bf16*)p;            p += MH * 2;
  __bf16* ql = (__bf16*)p;            p += MH * 2;
  __bf16* kh = (__bf16*)p;            p += MH * 2;
  __bf16* kl = (__bf16*)p;            p += MH * 2;
  __bf16* vth = (__bf16*)p;           p += MH * 2;
  __bf16* vtl = (__bf16*)p;           p += MH * 2;
  float* colmax = (float*)p;          p += (size_t)Bc * Tc * 4;
  float* colsum = (float*)p;          p += (size_t)Bc * Tc * 4;
  __bf16* wth = (__bf16*)p;           p += (size_t)3 * Cc * Hc * 2;
  __bf16* wtl = (__bf16*)p;

  prep_w<<<(3 * Cc * Hc + 255) / 256, 256, 0, stream>>>(Wq, Wk, Wv, wth, wtl);
  qkv_proj<<<128, 256, 0, stream>>>(x, wth, wtl, qh, ql, kh, kl, vth, vtl);
  col_stats<<<128, 256, 0, stream>>>(qh, ql, kh, kl, colmax, colsum);
  attn_out<<<128, 256, 0, stream>>>(qh, ql, kh, kl, vth, vtl, colmax, colsum,
                                    out);
}
